// MultiHeadAttention_23227183137318
// MI455X (gfx1250) — compile-verified, matrix operand order checked by reading, not run on hardware
//
#include <hip/hip_runtime.h>
#include <hip/hip_bf16.h>
#include <cstdint>
#include <cstddef>

// ---------------- problem sizes ----------------
#define B_  4
#define T_  2048
#define C_  2048
#define H_  16
#define HD_ 128
#define M_  (B_ * T_)                 // 8192 rows of x
static constexpr size_t BTC = (size_t)B_ * T_ * C_;   // 16,777,216
static constexpr size_t CCn = (size_t)C_ * C_;        // 4,194,304

// ---------------- WMMA types (CDNA5, wave32) ----------------
typedef __attribute__((ext_vector_type(16))) __bf16 v16bf;
typedef __attribute__((ext_vector_type(8)))  __bf16 v8bf;
typedef __attribute__((ext_vector_type(8)))  float  v8f;

__device__ __forceinline__ v8f wmma_bf16(v16bf a, v16bf b, v8f c) {
  // D(16x16,f32) = A(16x32,bf16) * B(32x16,bf16) + C
  return __builtin_amdgcn_wmma_f32_16x16x32_bf16(
      /*neg_a=*/false, a, /*neg_b=*/false, b,
      /*c_mod=*/(short)0, c, /*reuse_a=*/false, /*reuse_b=*/false);
}

// Load a 16x32 bf16 fragment from a row-major matrix (row stride `ld` elems).
// Per ISA layout: lanes 0-15 hold row r0+lane with K = k0+{0..7, 16..23};
// lanes 16-31 hold row r0+lane-16 with K = k0+{8..15, 24..31}.
// Used for A (rows of X / Q / P) and, by transpose symmetry, for B where
// "rows" are the N dimension (rows of W, rows of K, rows of vT).
__device__ __forceinline__ v16bf load_frag_rm(const __bf16* __restrict__ base,
                                              int ld, int r0, int k0) {
  const int lane = threadIdx.x & 31;
  const int r  = r0 + (lane & 15);
  const int kb = k0 + ((lane >> 4) << 3);
  const __bf16* p = base + (size_t)r * ld + kb;
  v8bf lo = *reinterpret_cast<const v8bf*>(p);        // K = kb .. kb+7
  v8bf hi = *reinterpret_cast<const v8bf*>(p + 16);   // K = kb+16 .. kb+23
  v16bf f;
#pragma unroll
  for (int i = 0; i < 8; ++i) { f[i] = lo[i]; f[i + 8] = hi[i]; }
  return f;
}

// ---------------- elementwise converters ----------------
__global__ void f32_to_bf16_kernel(const float* __restrict__ in,
                                   __bf16* __restrict__ out, size_t n) {
  size_t i = (size_t)blockIdx.x * blockDim.x + threadIdx.x;
  if (i < n) out[i] = (__bf16)in[i];
}

// RoPE on full C dim (half = C/2), f32 in -> bf16 out. One thread per pair.
__global__ void rope_kernel(const float* __restrict__ in,
                            const float* __restrict__ cosb,
                            const float* __restrict__ sinb,
                            __bf16* __restrict__ out) {
  size_t idx = (size_t)blockIdx.x * blockDim.x + threadIdx.x;
  const size_t npairs = BTC / 2;
  if (idx >= npairs) return;
  const int half = C_ / 2;
  int c      = (int)(idx % half);
  size_t row = idx / half;              // b*T + t
  int t      = (int)(row % T_);
  const float* r = in + row * C_;
  float x0 = r[c], x1 = r[c + half];
  float c0 = cosb[(size_t)t * C_ + c];
  float s0 = sinb[(size_t)t * C_ + c];
  float c1 = cosb[(size_t)t * C_ + c + half];
  float s1 = sinb[(size_t)t * C_ + c + half];
  out[row * C_ + c]        = (__bf16)(x0 * c0 - x1 * s0);
  out[row * C_ + c + half] = (__bf16)(x1 * c1 + x0 * s1);
}

// v [B,T,C] f32 -> vT [B*H, HD, T] bf16 (so P@V B-frags are contiguous in t)
__global__ void vtrans_kernel(const float* __restrict__ v,
                              __bf16* __restrict__ vT) {
  size_t idx = (size_t)blockIdx.x * blockDim.x + threadIdx.x;
  if (idx >= BTC) return;
  int c = (int)(idx % C_);
  size_t bt = idx / C_;
  int t = (int)(bt % T_);
  int b = (int)(bt / T_);
  int h = c / HD_, d = c % HD_;
  vT[(((size_t)(b * H_ + h) * HD_) + d) * T_ + t] = (__bf16)v[idx];
}

// ---------------- fused QKV projection GEMM ----------------
// grid(M/64, C/128), 256 threads (8 waves). Wave w: 64(M) x 16(N) tile,
// shared A-frags across the three weight matrices -> 12 WMMAs / K-step.
__global__ __launch_bounds__(256) void qkv_gemm_kernel(
    const __bf16* __restrict__ xb,
    const __bf16* __restrict__ wq, const __bf16* __restrict__ wk,
    const __bf16* __restrict__ wv,
    const float* __restrict__ bq, const float* __restrict__ bk,
    const float* __restrict__ bv,
    float* __restrict__ qf, float* __restrict__ kf, float* __restrict__ vf) {
  const int w    = threadIdx.x >> 5;
  const int lane = threadIdx.x & 31;
  const int m0 = blockIdx.x * 64;
  const int n0 = blockIdx.y * 128 + w * 16;

  v8f aq[4] = {}, ak[4] = {}, av[4] = {};
  for (int k0 = 0; k0 < C_; k0 += 32) {
    if (k0 + 32 < C_)   // keep the read stream ahead (global_prefetch_b8)
      __builtin_prefetch(xb + (size_t)m0 * C_ + k0 + 32, 0, 3);
    v16bf bfq = load_frag_rm(wq, C_, n0, k0);
    v16bf bfk = load_frag_rm(wk, C_, n0, k0);
    v16bf bfv = load_frag_rm(wv, C_, n0, k0);
#pragma unroll
    for (int i = 0; i < 4; ++i) {
      v16bf a = load_frag_rm(xb, C_, m0 + 16 * i, k0);
      aq[i] = wmma_bf16(a, bfq, aq[i]);
      ak[i] = wmma_bf16(a, bfk, ak[i]);
      av[i] = wmma_bf16(a, bfv, av[i]);
    }
  }
  const int col   = n0 + (lane & 15);
  const int rbase = (lane >> 4) << 3;
  const float biq = bq[col], bik = bk[col], biv = bv[col];
#pragma unroll
  for (int i = 0; i < 4; ++i) {
#pragma unroll
    for (int r = 0; r < 8; ++r) {
      size_t row = (size_t)(m0 + 16 * i + rbase + r);
      qf[row * C_ + col] = aq[i][r] + biq;
      kf[row * C_ + col] = ak[i][r] + bik;
      vf[row * C_ + col] = av[i][r] + biv;
    }
  }
}

// ---------------- causal flash attention ----------------
// grid(T/128, B*H), 256 threads. Wave w: 16 q rows, full HD=128 output,
// online softmax in registers, P re-layout via per-wave LDS tile.
__global__ __launch_bounds__(256) void attn_kernel(
    const __bf16* __restrict__ qr, const __bf16* __restrict__ kr,
    const __bf16* __restrict__ vT, __bf16* __restrict__ attn) {
  __shared__ __bf16 ptile[8][16][40];   // per-wave 16x32 P tile, padded
  const int w    = threadIdx.x >> 5;
  const int lane = threadIdx.x & 31;
  const int bh = blockIdx.y;            // b*H + h
  const int b  = bh / H_;
  const int h  = bh % H_;
  const int q0 = blockIdx.x * 128 + w * 16;

  const __bf16* qbase = qr + ((size_t)b * T_) * C_ + h * HD_;
  const __bf16* kbase = kr + ((size_t)b * T_) * C_ + h * HD_;
  const __bf16* vbase = vT + ((size_t)bh * HD_) * T_;

  v16bf qa[4];
#pragma unroll
  for (int c = 0; c < 4; ++c) qa[c] = load_frag_rm(qbase, C_, q0, c * 32);

  v8f o[8] = {};
  float mrow[8], lrow[8];
#pragma unroll
  for (int r = 0; r < 8; ++r) { mrow[r] = -3.0e38f; lrow[r] = 0.0f; }

  const float scale = 0.08838834764831845f;   // 1/sqrt(128)
  const int hb   = (lane >> 4) << 3;          // +8 rows for upper half-wave
  const int qhi  = q0 + 15;

  for (int j0 = 0; j0 <= qhi; j0 += 32) {
    v8f sa = {}, sb = {};
#pragma unroll
    for (int c = 0; c < 4; ++c) {
      v16bf ka = load_frag_rm(kbase, C_, j0,      c * 32);
      v16bf kb = load_frag_rm(kbase, C_, j0 + 16, c * 32);
      sa = wmma_bf16(qa[c], ka, sa);
      sb = wmma_bf16(qa[c], kb, sb);
    }
    // scale + causal mask
    const int colA = j0 + (lane & 15);
    const int colB = colA + 16;
    const bool need_mask = (j0 + 31) > q0;
#pragma unroll
    for (int r = 0; r < 8; ++r) {
      float x0 = sa[r] * scale, x1 = sb[r] * scale;
      if (need_mask) {
        int row = q0 + hb + r;
        if (colA > row) x0 = -3.0e38f;
        if (colB > row) x1 = -3.0e38f;
      }
      sa[r] = x0; sb[r] = x1;
    }
    // online softmax; row data lives across a 16-lane half at VGPR r
#pragma unroll
    for (int r = 0; r < 8; ++r) {
      float mx = fmaxf(sa[r], sb[r]);
#pragma unroll
      for (int msk = 1; msk < 16; msk <<= 1) mx = fmaxf(mx, __shfl_xor(mx, msk));
      float mnew  = fmaxf(mrow[r], mx);
      float alpha = __expf(mrow[r] - mnew);
      float p0 = __expf(sa[r] - mnew);
      float p1 = __expf(sb[r] - mnew);
      float ps = p0 + p1;
#pragma unroll
      for (int msk = 1; msk < 16; msk <<= 1) ps += __shfl_xor(ps, msk);
      lrow[r] = lrow[r] * alpha + ps;
      mrow[r] = mnew;
#pragma unroll
      for (int f = 0; f < 8; ++f) o[f][r] *= alpha;
      const int rl = hb + r;
      ptile[w][rl][lane & 15]        = (__bf16)p0;
      ptile[w][rl][16 + (lane & 15)] = (__bf16)p1;
    }
    asm volatile("s_wait_dscnt 0" ::: "memory");   // same-wave LDS RAW fence
    // reshape P (C-frag layout) -> A-frag layout via LDS
    v16bf pfrag;
    {
      const int rr = lane & 15;
      const int kb = hb;
#pragma unroll
      for (int i = 0; i < 8; ++i) {
        pfrag[i]     = ptile[w][rr][kb + i];
        pfrag[i + 8] = ptile[w][rr][kb + 16 + i];
      }
    }
#pragma unroll
    for (int f = 0; f < 8; ++f) {
      v16bf vb = load_frag_rm(vbase, T_, f * 16, j0);
      o[f] = wmma_bf16(pfrag, vb, o[f]);
    }
  }
  // epilogue: normalize and store bf16 attention output [B,T,C]
  const int colO = lane & 15;
#pragma unroll
  for (int f = 0; f < 8; ++f) {
#pragma unroll
    for (int r = 0; r < 8; ++r) {
      size_t trow = (size_t)(q0 + hb + r);
      float val = o[f][r] / lrow[r];
      attn[((size_t)b * T_ + trow) * C_ + h * HD_ + f * 16 + colO] = (__bf16)val;
    }
  }
}

// ---------------- output projection GEMM ----------------
__global__ __launch_bounds__(256) void out_gemm_kernel(
    const __bf16* __restrict__ ab, const __bf16* __restrict__ wo,
    const float* __restrict__ bo, float* __restrict__ out) {
  const int w    = threadIdx.x >> 5;
  const int lane = threadIdx.x & 31;
  const int m0 = blockIdx.x * 64;
  const int n0 = blockIdx.y * 128 + w * 16;
  v8f acc[4] = {};
  for (int k0 = 0; k0 < C_; k0 += 32) {
    if (k0 + 32 < C_)
      __builtin_prefetch(ab + (size_t)m0 * C_ + k0 + 32, 0, 3);
    v16bf bf = load_frag_rm(wo, C_, n0, k0);
#pragma unroll
    for (int i = 0; i < 4; ++i) {
      v16bf a = load_frag_rm(ab, C_, m0 + 16 * i, k0);
      acc[i] = wmma_bf16(a, bf, acc[i]);
    }
  }
  const int col   = n0 + (lane & 15);
  const int rbase = (lane >> 4) << 3;
  const float bias = bo[col];
#pragma unroll
  for (int i = 0; i < 4; ++i)
#pragma unroll
    for (int r = 0; r < 8; ++r)
      out[(size_t)(m0 + 16 * i + rbase + r) * C_ + col] = acc[i][r] + bias;
}

// ---------------- host launcher ----------------
extern "C" void kernel_launch(void* const* d_in, const int* in_sizes, int n_in,
                              void* d_out, int out_size, void* d_ws, size_t ws_size,
                              hipStream_t stream) {
  (void)in_sizes; (void)n_in; (void)out_size; (void)ws_size;
  const float* x    = (const float*)d_in[0];
  const float* cosb = (const float*)d_in[1];
  const float* sinb = (const float*)d_in[2];
  const float* Wq   = (const float*)d_in[3];
  const float* bq   = (const float*)d_in[4];
  const float* Wk   = (const float*)d_in[5];
  const float* bk   = (const float*)d_in[6];
  const float* Wv   = (const float*)d_in[7];
  const float* bv   = (const float*)d_in[8];
  const float* Wo   = (const float*)d_in[9];
  const float* bo   = (const float*)d_in[10];

  float* out   = (float*)d_out;          // [0, BTC): attention output
  float* k_new = out + BTC;              // pre-RoPE k
  float* v_new = out + 2 * BTC;          // v

  // workspace layout
  char* ws = (char*)d_ws;
  size_t off = 0;
  __bf16* xb  = (__bf16*)(ws + off); off += BTC * 2;
  __bf16* wqb = (__bf16*)(ws + off); off += CCn * 2;
  __bf16* wkb = (__bf16*)(ws + off); off += CCn * 2;
  __bf16* wvb = (__bf16*)(ws + off); off += CCn * 2;
  __bf16* wob = (__bf16*)(ws + off); off += CCn * 2;
  float*  qf  = (float*) (ws + off); off += BTC * 4;   // raw q (pre-RoPE)
  __bf16* qrb = (__bf16*)(ws + off); off += BTC * 2;   // q after RoPE
  __bf16* krb = (__bf16*)(ws + off); off += BTC * 2;   // k after RoPE
  __bf16* vTb = (__bf16*)(ws + off); off += BTC * 2;   // v transposed [BH,HD,T]
  __bf16* atb = (__bf16*)(ws + off); off += BTC * 2;   // attention out (bf16)

  const int thr = 256;
  // 1) fp32 -> bf16 converts
  f32_to_bf16_kernel<<<(int)((BTC + thr - 1) / thr), thr, 0, stream>>>(x,  xb,  BTC);
  f32_to_bf16_kernel<<<(int)((CCn + thr - 1) / thr), thr, 0, stream>>>(Wq, wqb, CCn);
  f32_to_bf16_kernel<<<(int)((CCn + thr - 1) / thr), thr, 0, stream>>>(Wk, wkb, CCn);
  f32_to_bf16_kernel<<<(int)((CCn + thr - 1) / thr), thr, 0, stream>>>(Wv, wvb, CCn);
  f32_to_bf16_kernel<<<(int)((CCn + thr - 1) / thr), thr, 0, stream>>>(Wo, wob, CCn);

  // 2) fused QKV projection (k_new / v_new written directly to d_out)
  qkv_gemm_kernel<<<dim3(M_ / 64, C_ / 128), thr, 0, stream>>>(
      xb, wqb, wkb, wvb, bq, bk, bv, qf, k_new, v_new);

  // 3) RoPE (q, k) + v transpose
  rope_kernel<<<(int)((BTC / 2 + thr - 1) / thr), thr, 0, stream>>>(qf,    cosb, sinb, qrb);
  rope_kernel<<<(int)((BTC / 2 + thr - 1) / thr), thr, 0, stream>>>(k_new, cosb, sinb, krb);
  vtrans_kernel<<<(int)((BTC + thr - 1) / thr), thr, 0, stream>>>(v_new, vTb);

  // 4) causal flash attention
  attn_kernel<<<dim3(T_ / 128, B_ * H_), thr, 0, stream>>>(qrb, krb, vTb, atb);

  // 5) output projection
  out_gemm_kernel<<<dim3(M_ / 64, C_ / 128), thr, 0, stream>>>(atb, wob, bo, out);
}
